// MultiHeadAttention_59287728554678
// MI455X (gfx1250) — matrix operand order repaired and checked by reading, not compiled, not run on hardware
//
#include <hip/hip_runtime.h>
#include <hip/hip_bf16.h>
#include <math.h>

typedef __attribute__((ext_vector_type(16))) __bf16 v16bf;
typedef __attribute__((ext_vector_type(8)))  __bf16 v8bf;
typedef __attribute__((ext_vector_type(8)))  float  v8f;

// Problem constants (fixed by reference): B=8, C=256, H=W=32 -> N=1024, nh=8, dk=dv=32
#define NBATCH   8
#define CCH      256
#define NSEQ     1024
#define NHEADS   8
#define DHEAD    32
#define NPIX     8192          // NBATCH * NSEQ
#define QSCALE   0.17677669529663687f   // 1/sqrt(32)

// ---------------- WMMA helpers (gfx1250, wave32) ----------------

__device__ __forceinline__ v8f wmma_bf16(v16bf a, v16bf b, v8f c) {
  // v_wmma_f32_16x16x32_bf16 : D = A(16x32) * B(32x16) + C(16x16 f32)
  return __builtin_amdgcn_wmma_f32_16x16x32_bf16(false, a, false, b, (short)0, c,
                                                 false, false);
}

// A-matrix 16x32 bf16 tile load. Lane L holds row m=L%16, half
// h=L/16; element e -> k = e + 16h. `row` -> (m, k=0).
__device__ __forceinline__ v16bf load_a_16x32(const __bf16* __restrict__ row, int h) {
  v8bf lo = *(const v8bf*)(row + 16 * h);       // 16B
  v8bf hi = *(const v8bf*)(row + 16 * h + 8);   // 16B
  v16bf a;
#pragma unroll
  for (int i = 0; i < 8; ++i) { a[i] = lo[i]; a[i + 8] = hi[i]; }
  return a;
}

// B-matrix 32x16 bf16 tile load. Lane L holds column n=L%16, half h=L/16;
// element e -> k = e + 16h. `colp` -> (k=0, n) with K contiguous.
__device__ __forceinline__ v16bf load_b_32x16(const __bf16* __restrict__ colp, int h) {
  return *(const v16bf*)(colp + 16 * h);        // 32B
}

__device__ __forceinline__ unsigned pack_bf16x2(float lo, float hi) {
  union { __bf16 b[2]; unsigned u; } t;
  t.b[0] = (__bf16)lo; t.b[1] = (__bf16)hi;
  return t.u;
}

// ---------------- stage 0: pack inputs to bf16 ----------------

__global__ void pack_x_kernel(const float* __restrict__ x, __bf16* __restrict__ xT) {
  int idx = blockIdx.x * 256 + threadIdx.x;     // 0 .. NPIX*CCH-1
  int p = idx >> 8;
  int c = idx & 255;
  int b = p >> 10;
  int n = p & 1023;
  xT[idx] = (__bf16)x[(b * CCH + c) * NSEQ + n];
}

__global__ void cvt_f32_bf16_kernel(const float* __restrict__ s, __bf16* __restrict__ d, int n) {
  int i = blockIdx.x * 256 + threadIdx.x;
  if (i < n) d[i] = (__bf16)s[i];
}

// ---------------- stage 1: fused QKV projection GEMM ----------------
// One wave computes a 16(pixel) x 64(outchan) tile: A-operand reused across
// 4 B-operands -> 32 WMMAs per wave, 1.25KB loaded per WMMA.
// sel = blockIdx.z : 0->Q (scaled, [seq][d]), 1->K ([seq][d]), 2->V ([d][seq])
__global__ void qkv_proj_kernel(const __bf16* __restrict__ xT,
                                const __bf16* __restrict__ wq, const __bf16* __restrict__ wk,
                                const __bf16* __restrict__ wv,
                                const float* __restrict__ bq, const float* __restrict__ bk,
                                const float* __restrict__ bv,
                                __bf16* __restrict__ Qw, __bf16* __restrict__ Kw,
                                __bf16* __restrict__ Vt) {
  const int lane = threadIdx.x & 31;
  const int wave = blockIdx.x * 4 + (threadIdx.x >> 5);   // 0..2047
  const int sel  = blockIdx.z;
  const int mtile = wave >> 2;                            // 0..511 : 16-pixel tile
  const int ng    = wave & 3;                             // 4 x 16-outchan tiles
  const int lr = lane & 15;
  const int h  = lane >> 4;

  const __bf16* W    = (sel == 0) ? wq : ((sel == 1) ? wk : wv);
  const float*  bias = (sel == 0) ? bq : ((sel == 1) ? bk : bv);

  const int m0 = mtile * 16;
  int oj[4];
#pragma unroll
  for (int j = 0; j < 4; ++j) oj[j] = (ng * 4 + j) * 16 + lr;

  const __bf16* arow = xT + (size_t)(m0 + lr) * CCH;
  const __bf16* brow0 = W + (size_t)oj[0] * CCH;
  const __bf16* brow1 = W + (size_t)oj[1] * CCH;
  const __bf16* brow2 = W + (size_t)oj[2] * CCH;
  const __bf16* brow3 = W + (size_t)oj[3] * CCH;

  v8f acc[4] = {};
#pragma unroll
  for (int k0 = 0; k0 < CCH; k0 += 32) {
    v16bf a = load_a_16x32(arow + k0, h);
    acc[0] = wmma_bf16(a, load_b_32x16(brow0 + k0, h), acc[0]);
    acc[1] = wmma_bf16(a, load_b_32x16(brow1 + k0, h), acc[1]);
    acc[2] = wmma_bf16(a, load_b_32x16(brow2 + k0, h), acc[2]);
    acc[3] = wmma_bf16(a, load_b_32x16(brow3 + k0, h), acc[3]);
  }

#pragma unroll
  for (int j = 0; j < 4; ++j) {
    const float bsc = bias[oj[j]];
    const int hh = oj[j] >> 5;
    const int d  = oj[j] & 31;
    if (sel == 2) {
      // V transposed: Vt[((b*8+hh)*32 + d)*1024 + npix]; pixels contiguous
      int pix0 = m0 + 8 * h;
      int b_ = pix0 >> 10, np = pix0 & 1023;
      v8bf ov;
#pragma unroll
      for (int r = 0; r < 8; ++r) ov[r] = (__bf16)(acc[j][r] + bsc);
      *(v8bf*)(Vt + ((size_t)(b_ * NHEADS + hh) * DHEAD + d) * NSEQ + np) = ov;
    } else {
      __bf16* dst = (sel == 0) ? Qw : Kw;
      const float scale = (sel == 0) ? QSCALE : 1.0f;
#pragma unroll
      for (int r = 0; r < 8; ++r) {
        int pix = m0 + r + 8 * h;
        int b_ = pix >> 10, np = pix & 1023;
        dst[((size_t)(b_ * NHEADS + hh) * NSEQ + np) * DHEAD + d] =
            (__bf16)((acc[j][r] + bsc) * scale);
      }
    }
  }
}

// ---------------- stage 2: flash attention ----------------
// One wave handles FOUR 16-query tiles of one (b,h): each K/V A-tile load is
// reused by 4 WMMAs (16 WMMAs per 8 b128 loads per 32-key chunk).
// S^T formulation (A=K rows, B=Q cols) puts softmax stats one-query-per-lane.
__global__ void attn_kernel(const __bf16* __restrict__ Qw, const __bf16* __restrict__ Kw,
                            const __bf16* __restrict__ Vt, __bf16* __restrict__ Ow) {
  const int lane = threadIdx.x & 31;
  const int wave = blockIdx.x * 4 + (threadIdx.x >> 5);   // 0..1023
  const int qgrp = wave & 15;                             // 4 query tiles each
  const int bh   = wave >> 4;                             // 0..63 = b*8+h
  const int lr = lane & 15;
  const int h  = lane >> 4;

  const __bf16* Qbase = Qw + (size_t)bh * NSEQ * DHEAD;
  const __bf16* Kbase = Kw + (size_t)bh * NSEQ * DHEAD;
  const __bf16* Vbase = Vt + (size_t)bh * DHEAD * NSEQ;

  // Q as B-operands (k=d contiguous per query row), loop invariant
  v16bf bq[4];
#pragma unroll
  for (int qt = 0; qt < 4; ++qt)
    bq[qt] = *(const v16bf*)(Qbase + (size_t)((qgrp * 4 + qt) * 16 + lr) * DHEAD + 16 * h);

  v8f oacc[4][2] = {};
  float mstat[4], lstat[4];
#pragma unroll
  for (int qt = 0; qt < 4; ++qt) { mstat[qt] = -INFINITY; lstat[qt] = 0.0f; }

  for (int kc = 0; kc < NSEQ; kc += 32) {
    // shared K/V A-tiles for this 32-key chunk
    const __bf16* krow = Kbase + (size_t)(kc + lr) * DHEAD;
    v16bf ak0 = load_a_16x32(krow, h);
    v16bf ak1 = load_a_16x32(krow + 16 * DHEAD, h);
    v16bf av0 = load_a_16x32(Vbase + (size_t)lr * NSEQ + kc, h);
    v16bf av1 = load_a_16x32(Vbase + (size_t)(16 + lr) * NSEQ + kc, h);

#pragma unroll
    for (int qt = 0; qt < 4; ++qt) {
      v8f z = {};
      v8f s0 = wmma_bf16(ak0, bq[qt], z);       // D[key][query]
      v8f s1 = wmma_bf16(ak1, bq[qt], z);

      // online softmax stats: per query = per lane column, halves combined
      float cm = fmaxf(s0[0], s1[0]);
#pragma unroll
      for (int r = 1; r < 8; ++r) { cm = fmaxf(cm, s0[r]); cm = fmaxf(cm, s1[r]); }
      cm = fmaxf(cm, __shfl_xor(cm, 16, 32));
      const float mnew  = fmaxf(mstat[qt], cm);
      const float alpha = __expf(mstat[qt] - mnew);
      mstat[qt] = mnew;

      float p0[8], p1[8], ls = 0.0f;
#pragma unroll
      for (int r = 0; r < 8; ++r) {
        p0[r] = __expf(s0[r] - mnew);
        p1[r] = __expf(s1[r] - mnew);
        ls += p0[r] + p1[r];
      }
      ls += __shfl_xor(ls, 16, 32);
      lstat[qt] = lstat[qt] * alpha + ls;

      // Build P^T B-operand. VGPR j holds elements (2j,2j+1) -> k = 2j+16h.
      // Pack bf16 pairs first so each half-swap shuffle moves two values.
      union { unsigned u[8]; v16bf v; } bp;
#pragma unroll
      for (int j = 0; j < 4; ++j) {
        unsigned a0 = pack_bf16x2(p0[2 * j], p0[2 * j + 1]);   // own, tile0
        unsigned a1 = pack_bf16x2(p1[2 * j], p1[2 * j + 1]);   // own, tile1
        unsigned b0 = (unsigned)__shfl_xor((int)a0, 16, 32);   // partner, tile0
        unsigned b1 = (unsigned)__shfl_xor((int)a1, 16, 32);   // partner, tile1
        bp.u[j]     = h ? b1 : a0;
        bp.u[j + 4] = h ? a1 : b0;
      }

#pragma unroll
      for (int r = 0; r < 8; ++r) { oacc[qt][0][r] *= alpha; oacc[qt][1][r] *= alpha; }
      oacc[qt][0] = wmma_bf16(av0, bp.v, oacc[qt][0]);   // O[d=0..15][query]
      oacc[qt][1] = wmma_bf16(av1, bp.v, oacc[qt][1]);   // O[d=16..31][query]
    }
  }

  // normalize + store O as [pixel][nh*dv] bf16 rows for the output projection
  const int b_ = bh >> 3, hh = bh & 7;
#pragma unroll
  for (int qt = 0; qt < 4; ++qt) {
    const float inv = 1.0f / lstat[qt];
    const size_t pix = (size_t)b_ * NSEQ + (qgrp * 4 + qt) * 16 + lr;
    __bf16* obase = Ow + pix * CCH + hh * DHEAD;
    v8bf st;
#pragma unroll
    for (int r = 0; r < 8; ++r) st[r] = (__bf16)(oacc[qt][0][r] * inv);
    *(v8bf*)(obase + 8 * h) = st;                        // d = r + 8h
#pragma unroll
    for (int r = 0; r < 8; ++r) st[r] = (__bf16)(oacc[qt][1][r] * inv);
    *(v8bf*)(obase + 16 + 8 * h) = st;                   // d = 16 + r + 8h
  }
}

// ---------------- stage 3: output projection GEMM, fp32 out ----------------
// Same 16x64-per-wave blocking as stage 1.
__global__ void proj_o_kernel(const __bf16* __restrict__ Ow, const __bf16* __restrict__ wo,
                              const float* __restrict__ bo, float* __restrict__ out) {
  const int lane = threadIdx.x & 31;
  const int wave = blockIdx.x * 4 + (threadIdx.x >> 5);   // 0..2047
  const int mtile = wave >> 2;
  const int ng    = wave & 3;
  const int lr = lane & 15;
  const int h  = lane >> 4;

  const int m0 = mtile * 16;
  int oj[4];
#pragma unroll
  for (int j = 0; j < 4; ++j) oj[j] = (ng * 4 + j) * 16 + lr;

  const __bf16* arow = Ow + (size_t)(m0 + lr) * CCH;
  const __bf16* brow0 = wo + (size_t)oj[0] * CCH;
  const __bf16* brow1 = wo + (size_t)oj[1] * CCH;
  const __bf16* brow2 = wo + (size_t)oj[2] * CCH;
  const __bf16* brow3 = wo + (size_t)oj[3] * CCH;

  v8f acc[4] = {};
#pragma unroll
  for (int k0 = 0; k0 < CCH; k0 += 32) {
    v16bf a = load_a_16x32(arow + k0, h);
    acc[0] = wmma_bf16(a, load_b_32x16(brow0 + k0, h), acc[0]);
    acc[1] = wmma_bf16(a, load_b_32x16(brow1 + k0, h), acc[1]);
    acc[2] = wmma_bf16(a, load_b_32x16(brow2 + k0, h), acc[2]);
    acc[3] = wmma_bf16(a, load_b_32x16(brow3 + k0, h), acc[3]);
  }

  const int pix0 = m0 + 8 * h;
  const int b_ = pix0 >> 10, np = pix0 & 1023;
#pragma unroll
  for (int j = 0; j < 4; ++j) {
    const float bsc = bo[oj[j]];
    float* po = out + ((size_t)(b_ * CCH) + oj[j]) * NSEQ + np;   // pixels contiguous
#pragma unroll
    for (int r = 0; r < 8; ++r) po[r] = acc[j][r] + bsc;
  }
}

// ---------------- launch ----------------

extern "C" void kernel_launch(void* const* d_in, const int* in_sizes, int n_in,
                              void* d_out, int out_size, void* d_ws, size_t ws_size,
                              hipStream_t stream) {
  const float* x  = (const float*)d_in[0];
  const float* Wq = (const float*)d_in[1];
  const float* bq = (const float*)d_in[2];
  const float* Wk = (const float*)d_in[3];
  const float* bk = (const float*)d_in[4];
  const float* Wv = (const float*)d_in[5];
  const float* bv = (const float*)d_in[6];
  const float* Wo = (const float*)d_in[7];
  const float* bo = (const float*)d_in[8];
  float* out = (float*)d_out;

  // bf16 workspace layout (elements)
  __bf16* ws = (__bf16*)d_ws;
  __bf16* xT = ws;                        // NPIX*CCH (reused as Ow later)
  __bf16* wq = xT + (size_t)NPIX * CCH;
  __bf16* wk = wq + CCH * CCH;
  __bf16* wv = wk + CCH * CCH;
  __bf16* wo = wv + CCH * CCH;
  __bf16* Qw = wo + CCH * CCH;            // [b,h][seq][d]
  __bf16* Kw = Qw + (size_t)NPIX * DHEAD; // [b,h][seq][d]
  __bf16* Vt = Kw + (size_t)NPIX * DHEAD; // [b,h][d][seq]
  __bf16* Ow = xT;                        // alias: xT dead after projections

  pack_x_kernel<<<NPIX * CCH / 256, 256, 0, stream>>>(x, xT);
  cvt_f32_bf16_kernel<<<CCH * CCH / 256, 256, 0, stream>>>(Wq, wq, CCH * CCH);
  cvt_f32_bf16_kernel<<<CCH * CCH / 256, 256, 0, stream>>>(Wk, wk, CCH * CCH);
  cvt_f32_bf16_kernel<<<CCH * CCH / 256, 256, 0, stream>>>(Wv, wv, CCH * CCH);
  cvt_f32_bf16_kernel<<<CCH * CCH / 256, 256, 0, stream>>>(Wo, wo, CCH * CCH);

  dim3 gqkv(512, 1, 3);                   // 2048 waves per matrix, 4 waves/block
  qkv_proj_kernel<<<gqkv, 128, 0, stream>>>(xT, wq, wk, wv, bq, bk, bv, Qw, Kw, Vt);

  attn_kernel<<<256, 128, 0, stream>>>(Qw, Kw, Vt, Ow);   // 1024 waves

  proj_o_kernel<<<512, 128, 0, stream>>>(Ow, wo, bo, out);
}
